// LinkGNN_MLP_84825604096062
// MI455X (gfx1250) — compile-verified, matrix operand order checked by reading, not run on hardware
//
#include <hip/hip_runtime.h>
#include <hip/hip_bf16.h>

// CDNA5 / gfx1250 GNN link-prediction pipeline.
// wave32, WMMA 16x16x32 f16 (f32 accum). All matrix dims are exact multiples
// of the WMMA tile (50000=3125*16, 100000=6250*16, K in {128,64}, N=64).

typedef _Float16 f16;
typedef __attribute__((ext_vector_type(8)))  _Float16 v8h;
typedef __attribute__((ext_vector_type(16))) _Float16 v16h;
typedef __attribute__((ext_vector_type(8)))  float    v8f;

// ---------------------------------------------------------------- small utils

__global__ void k_fill1(float* p, int n) {
    int i = blockIdx.x * blockDim.x + threadIdx.x;
    if (i < n) p[i] = 1.0f;   // self-loop contributes 1 to in-degree
}

__global__ void k_deg(const int* __restrict__ dst, float* __restrict__ deg, int E) {
    int e = blockIdx.x * blockDim.x + threadIdx.x;
    if (e < E) unsafeAtomicAdd(&deg[dst[e]], 1.0f);   // global_atomic_add_f32
}

__global__ void k_rsqrt_inplace(float* p, int n) {
    int i = blockIdx.x * blockDim.x + threadIdx.x;
    if (i < n) p[i] = rsqrtf(p[i]);                   // deg >= 1 always
}

__global__ void k_cvt_f16(const float* __restrict__ a, f16* __restrict__ b, int n) {
    int i = blockIdx.x * blockDim.x + threadIdx.x;
    if (i < n) b[i] = (f16)a[i];
}

// Pack a row-major KxN f32 weight into per-fragment B layout:
// frag f = kt*nTiles + nt; within a frag: 32 lanes x 16 halves contiguous.
// Lane L (L<16) holds col n=nt*16+L, halves i: k = kt*32 + (i<8 ? i : i+8);
// lanes 16..31 add +8 to k (ISA 7.12.2 16-bit operand layout, B mirrored from A).
__global__ void k_packB(const float* __restrict__ W, f16* __restrict__ out,
                        int K, int nTiles) {
    int id = blockIdx.x * blockDim.x + threadIdx.x;
    int total = K * nTiles * 16;
    if (id >= total) return;
    int f    = id >> 9;          // 512 halves per fragment
    int rem  = id & 511;
    int lane = rem >> 4;
    int i    = rem & 15;
    int nt = f % nTiles;
    int kt = f / nTiles;
    int hi = lane >> 4;
    int k = kt * 32 + (i < 8 ? i : i + 8) + hi * 8;
    int n = nt * 16 + (lane & 15);
    out[id] = (f16)W[(long)k * (nTiles * 16) + n];
}

// ------------------------------------------------------------------ WMMA GEMM
// C[M x (nTiles*16)] = A[M x K] (f16 row-major) * Bfrag (pre-packed f16).
// One wave per 16x16 output tile; kTiles accumulating WMMAs.
__global__ void k_gemm(const f16* __restrict__ A, const f16* __restrict__ Bf,
                       float* __restrict__ C, int Mtiles, int kTiles,
                       int nTiles, int K) {
    int wid  = threadIdx.x >> 5;
    int lane = threadIdx.x & 31;
    int tile = blockIdx.x * 8 + wid;
    if (tile >= Mtiles * nTiles) return;           // wave-uniform guard
    int nt = tile % nTiles, mt = tile / nTiles;
    int l16 = lane & 15, hi = lane >> 4;

    const f16* arow = A + (long)(mt * 16 + l16) * K + hi * 8;
    v8f c = {};
    for (int kt = 0; kt < kTiles; ++kt) {
        const f16* ap = arow + kt * 32;
        v8h lo  = *(const v8h*)ap;                  // k: +0..7   (lane-half adj.)
        v8h hi8 = *(const v8h*)(ap + 16);           // k: +16..23
        v16h a = __builtin_shufflevector(lo, hi8, 0,1,2,3,4,5,6,7,
                                                  8,9,10,11,12,13,14,15);
        v16h b = *(const v16h*)(Bf + ((long)(kt * nTiles + nt) * 32 + lane) * 16);
        c = __builtin_amdgcn_wmma_f32_16x16x32_f16(false, a, false, b,
                                                   (short)0, c, false, false);
    }
    int N = nTiles * 16;
    float* crow = C + (long)(mt * 16 + hi * 8) * N + nt * 16 + l16;
#pragma unroll
    for (int r = 0; r < 8; ++r) crow[(long)r * N] = c[r];
}

// ------------------------------------------------------- GCN aggregation step

// agg[i][c] = t[i][c] * dinv[i]^2   (self-loop term)
__global__ void k_selfinit(const float* __restrict__ t, const float* __restrict__ dinv,
                           float* __restrict__ agg, int n) {
    int i = blockIdx.x * blockDim.x + threadIdx.x;
    if (i < n) {
        float d = dinv[i >> 6];
        agg[i] = t[i] * d * d;
    }
}

// one wave per edge: agg[dst] += t[src] * dinv[src]*dinv[dst]  (64 f32, 2/lane)
__global__ void k_scatter(const int* __restrict__ src, const int* __restrict__ dst,
                          const float* __restrict__ dinv, const float* __restrict__ t,
                          float* __restrict__ agg, int E) {
    int wid  = threadIdx.x >> 5;
    int lane = threadIdx.x & 31;
    int e = blockIdx.x * 8 + wid;
    if (e >= E) return;
    int s = src[e], d = dst[e];
    float nrm = dinv[s] * dinv[d];
    const float* trow = t + (long)s * 64;
    float* arow = agg + (long)d * 64;
    unsafeAtomicAdd(&arow[lane],      trow[lane]      * nrm);
    unsafeAtomicAdd(&arow[lane + 32], trow[lane + 32] * nrm);
}

// h = f16(relu(agg + b))
__global__ void k_bias_relu_cvt(const float* __restrict__ agg, const float* __restrict__ b,
                                f16* __restrict__ out, int n) {
    int i = blockIdx.x * blockDim.x + threadIdx.x;
    if (i < n) {
        float v = agg[i] + b[i & 63];
        out[i] = (f16)(v > 0.f ? v : 0.f);
    }
}

// z = agg + b (in place, stays f32 for the decoder gather)
__global__ void k_bias_inplace(float* __restrict__ agg, const float* __restrict__ b, int n) {
    int i = blockIdx.x * blockDim.x + threadIdx.x;
    if (i < n) agg[i] += b[i & 63];
}

// --------------------------------------------------------------- fused decode
// per wave: 16 edge pairs. ef = z[a]*z[b] built directly as an A fragment in
// registers; WMMA x P1 -> relu -> LDS -> A frag -> WMMA x P2 -> relu -> LDS ->
// 64-wide dot with P3 (shfl_xor reduction across the two lane halves).
__global__ void k_decode(const float* __restrict__ z,
                         const int* __restrict__ eA, const int* __restrict__ eB,
                         const f16* __restrict__ p1f, const float* __restrict__ pb1,
                         const f16* __restrict__ p2f, const float* __restrict__ pb2,
                         const float* __restrict__ P3, const float* __restrict__ pb3,
                         float* __restrict__ out, int tiles) {
    __shared__ __align__(32) f16 lds[8 * 2048];     // 2 x (16x64) halves / wave
    int wid  = threadIdx.x >> 5;
    int lane = threadIdx.x & 31;
    int tile = blockIdx.x * 8 + wid;
    if (tile >= tiles) return;                       // wave-uniform
    int l16 = lane & 15, hi = lane >> 4;
    int row = tile * 16 + l16;

    const float* za = z + (long)eA[row] * 64;
    const float* zb = z + (long)eB[row] * 64;
    f16* hb1 = lds + wid * 2048;
    f16* hb2 = hb1 + 1024;

    // ---- layer 1: ef @ P1
    v8f c[4] = {{}, {}, {}, {}};
    for (int kt = 0; kt < 2; ++kt) {
        int koff = kt * 32 + hi * 8;
        v16h a;
#pragma unroll
        for (int i = 0; i < 8; ++i) {
            a[i]     = (f16)(za[koff + i]      * zb[koff + i]);
            a[i + 8] = (f16)(za[koff + 16 + i] * zb[koff + 16 + i]);
        }
#pragma unroll
        for (int nt = 0; nt < 4; ++nt) {
            v16h b = *(const v16h*)(p1f + ((long)(kt * 4 + nt) * 32 + lane) * 16);
            c[nt] = __builtin_amdgcn_wmma_f32_16x16x32_f16(false, a, false, b,
                                                           (short)0, c[nt], false, false);
        }
    }
#pragma unroll
    for (int nt = 0; nt < 4; ++nt) {
        int col = nt * 16 + l16;
        float bv = pb1[col];
#pragma unroll
        for (int r = 0; r < 8; ++r) {
            float v = c[nt][r] + bv;
            hb1[(r + hi * 8) * 64 + col] = (f16)(v > 0.f ? v : 0.f);
        }
    }
    asm volatile("s_wait_dscnt 0" ::: "memory");     // wave-local LDS visibility

    // ---- layer 2: h @ P2
    v8f d[4] = {{}, {}, {}, {}};
    for (int kt = 0; kt < 2; ++kt) {
        const f16* ap = hb1 + l16 * 64 + kt * 32 + hi * 8;
        v8h lo  = *(const v8h*)ap;
        v8h hi8 = *(const v8h*)(ap + 16);
        v16h a = __builtin_shufflevector(lo, hi8, 0,1,2,3,4,5,6,7,
                                                  8,9,10,11,12,13,14,15);
#pragma unroll
        for (int nt = 0; nt < 4; ++nt) {
            v16h b = *(const v16h*)(p2f + ((long)(kt * 4 + nt) * 32 + lane) * 16);
            d[nt] = __builtin_amdgcn_wmma_f32_16x16x32_f16(false, a, false, b,
                                                           (short)0, d[nt], false, false);
        }
    }
#pragma unroll
    for (int nt = 0; nt < 4; ++nt) {
        int col = nt * 16 + l16;
        float bv = pb2[col];
#pragma unroll
        for (int r = 0; r < 8; ++r) {
            float v = d[nt][r] + bv;
            hb2[(r + hi * 8) * 64 + col] = (f16)(v > 0.f ? v : 0.f);
        }
    }
    asm volatile("s_wait_dscnt 0" ::: "memory");

    // ---- layer 3: dot with P3 (each lane half sums 32, combine across halves)
    const f16* hp = hb2 + l16 * 64 + hi * 32;
    float s = 0.f;
#pragma unroll
    for (int i = 0; i < 32; ++i) s += (float)hp[i] * P3[hi * 32 + i];
    s += __shfl_xor(s, 16, 32);
    if (hi == 0) out[row] = s + pb3[0];
}

// -------------------------------------------------------------------- launch

extern "C" void kernel_launch(void* const* d_in, const int* in_sizes, int n_in,
                              void* d_out, int out_size, void* d_ws, size_t ws_size,
                              hipStream_t stream) {
    const float* x   = (const float*)d_in[0];
    const int*   ei  = (const int*)  d_in[1];   // [2,E] row-major
    const int*   eli = (const int*)  d_in[2];   // [2,EL]
    const float* W1  = (const float*)d_in[3];
    const float* b1  = (const float*)d_in[4];
    const float* W2  = (const float*)d_in[5];
    const float* b2  = (const float*)d_in[6];
    const float* P1  = (const float*)d_in[7];
    const float* pb1 = (const float*)d_in[8];
    const float* P2  = (const float*)d_in[9];
    const float* pb2 = (const float*)d_in[10];
    const float* P3  = (const float*)d_in[11];
    const float* pb3 = (const float*)d_in[12];
    float* out = (float*)d_out;

    const int N  = in_sizes[0] / 128;   // 50000
    const int E  = in_sizes[1] / 2;     // 800000
    const int EL = in_sizes[2] / 2;     // 100000

    // workspace carve-out (256B aligned)
    char* w = (char*)d_ws;
    size_t off = 0;
    auto carve = [&](size_t bytes) -> void* {
        void* p = w + off;
        off = (off + bytes + 255) & ~(size_t)255;
        return p;
    };
    float* deg  = (float*)carve((size_t)N * 4);            // becomes dinv
    f16*   xh   = (f16*)  carve((size_t)N * 128 * 2);
    f16*   w1f  = (f16*)  carve(128 * 64 * 2);
    f16*   w2f  = (f16*)  carve(64 * 64 * 2);
    f16*   p1f  = (f16*)  carve(64 * 64 * 2);
    f16*   p2f  = (f16*)  carve(64 * 64 * 2);
    float* t    = (float*)carve((size_t)N * 64 * 4);       // GEMM out (both layers)
    float* agg  = (float*)carve((size_t)N * 64 * 4);       // aggregated / z
    f16*   h1h  = (f16*)  carve((size_t)N * 64 * 2);
    (void)ws_size; (void)n_in; (void)out_size;

    const int B = 256;
    const int Mtiles = N / 16;          // 3125 (exact)
    const int gemmBlocks = (Mtiles * 4 + 7) / 8;

    // degree -> dinv
    k_fill1<<<(N + B - 1) / B, B, 0, stream>>>(deg, N);
    k_deg<<<(E + B - 1) / B, B, 0, stream>>>(ei + E, deg, E);
    k_rsqrt_inplace<<<(N + B - 1) / B, B, 0, stream>>>(deg, N);

    // f16 conversions / weight packing
    k_cvt_f16<<<((size_t)N * 128 + B - 1) / B, B, 0, stream>>>(x, xh, N * 128);
    k_packB<<<(128 * 64 + B - 1) / B, B, 0, stream>>>(W1, w1f, 128, 4);
    k_packB<<<(64 * 64 + B - 1) / B, B, 0, stream>>>(W2, w2f, 64, 4);
    k_packB<<<(64 * 64 + B - 1) / B, B, 0, stream>>>(P1, p1f, 64, 4);
    k_packB<<<(64 * 64 + B - 1) / B, B, 0, stream>>>(P2, p2f, 64, 4);

    // layer 1: t = x @ W1 ; aggregate ; relu(+b1) -> h1h (f16)
    k_gemm<<<gemmBlocks, B, 0, stream>>>(xh, w1f, t, Mtiles, 4, 4, 128);
    k_selfinit<<<((size_t)N * 64 + B - 1) / B, B, 0, stream>>>(t, deg, agg, N * 64);
    k_scatter<<<(E + 7) / 8, B, 0, stream>>>(ei, ei + E, deg, t, agg, E);
    k_bias_relu_cvt<<<((size_t)N * 64 + B - 1) / B, B, 0, stream>>>(agg, b1, h1h, N * 64);

    // layer 2: t = h1 @ W2 ; aggregate ; +b2 -> z (f32, in agg)
    k_gemm<<<gemmBlocks, B, 0, stream>>>(h1h, w2f, t, Mtiles, 2, 4, 64);
    k_selfinit<<<((size_t)N * 64 + B - 1) / B, B, 0, stream>>>(t, deg, agg, N * 64);
    k_scatter<<<(E + 7) / 8, B, 0, stream>>>(ei, ei + E, deg, t, agg, E);
    k_bias_inplace<<<((size_t)N * 64 + B - 1) / B, B, 0, stream>>>(agg, b2, N * 64);

    // fused decoder
    const int tiles = EL / 16;          // 6250 (exact)
    k_decode<<<(tiles + 7) / 8, B, 0, stream>>>(agg, eli, eli + EL,
                                                p1f, pb1, p2f, pb2, P3, pb3,
                                                out, tiles);
}
